// Model_35940286333312
// MI455X (gfx1250) — matrix operand order repaired and checked by reading, not compiled, not run on hardware
//
#include <hip/hip_runtime.h>
#include <hip/hip_bf16.h>

#define NN 50000
#define HH 5000
#define EE 400000
#define BB 2
#define CC 64
#define BC (BB*CC)          // 128 floats per node row in x1 layout

typedef _Float16 f16;
typedef __attribute__((ext_vector_type(16))) _Float16 v16h;
typedef __attribute__((ext_vector_type(8)))  _Float16 v8h;
typedef __attribute__((ext_vector_type(8)))  float    v8f;

// ---------------------------------------------------------------- utilities
__global__ void k_zero(float* __restrict__ p, int n) {
    int i = blockIdx.x * blockDim.x + threadIdx.x;
    if (i < n) p[i] = 0.0f;
}

// order-preserving float <-> uint key for atomicMax-based float max
__device__ __forceinline__ unsigned encf(float f) {
    unsigned u = __float_as_uint(f);
    return (u & 0x80000000u) ? ~u : (u | 0x80000000u);
}
__device__ __forceinline__ float decf(unsigned u) {
    return (u & 0x80000000u) ? __uint_as_float(u & 0x7FFFFFFFu)
                             : __uint_as_float(~u);
}

// binary search for [start,end) of hyperedge h in sorted hedge_idx
__device__ __forceinline__ void edge_range(const int* __restrict__ hedge_idx,
                                           int h, int& start, int& end) {
    int lo = 0, hi = EE;
    while (lo < hi) { int mid = (lo + hi) >> 1; if (hedge_idx[mid] <  h)     lo = mid + 1; else hi = mid; }
    start = lo;
    lo = start; hi = EE;
    while (lo < hi) { int mid = (lo + hi) >> 1; if (hedge_idx[mid] <  h + 1) lo = mid + 1; else hi = mid; }
    end = lo;
}

// ---------------------------------------------------------------- 1) GEMM via WMMA
// x[b, n, k] (f32) @ weight[k, d] (f32) -> x1[n, b, d] (f32), node-major so
// downstream gathers read one contiguous 512B row per incidence entry.
// Block = 128 threads = 4 wave32; each wave owns a 16x16 output tile, K=64 in
// two v_wmma_f32_16x16x32_f16 steps. Weight is staged TRANSPOSED in LDS so the
// B-fragment (per-lane element e <-> K = 16*half + k0 + e) is one contiguous
// 32B read; the A-fragment is one contiguous 32B read.
__global__ void k_gemm(const float* __restrict__ x, const float* __restrict__ w,
                       float* __restrict__ x1) {
    __shared__ f16 sWt[64 * 64];  // weight^T: sWt[d*64 + k]
    __shared__ f16 sX[16 * 64];   // 16 node rows: sX[m*64 + k]
    const int b     = blockIdx.y;
    const int nbase = blockIdx.x * 16;
    const int tid   = threadIdx.x;

    for (int i = tid; i < 64 * 64; i += 128) {
        int k = i >> 6, d = i & 63;
        sWt[d * 64 + k] = (f16)w[i];
    }
    const float* xp = x + ((size_t)b * NN + nbase) * CC;
    for (int i = tid; i < 16 * 64; i += 128) sX[i] = (f16)xp[i];
    __syncthreads();

    const int wv   = tid >> 5;          // wave id -> column tile (16*wv)
    const int lane = tid & 31;
    const int half = lane >> 4;         // lane group 0-15 / 16-31
    const int mrow = lane & 15;
    const int ncol = (lane & 15) + wv * 16;

    v8f acc = {};
    for (int k0 = 0; k0 < 64; k0 += 32) {
        // A 16x32 (MxK): lane element e <-> K = 16*half+k0+e
        // -> one contiguous v16h load (matches B).
        const f16* ap = &sX[mrow * 64 + 16 * half + k0];
        v16h afrag = *(const v16h*)(ap);
        // B 32x16 (KxN): lane column n = lane&15, element e <-> K = 16*half+k0+e
        // -> one contiguous v16h load from transposed weight.
        v16h bfrag = *(const v16h*)&sWt[ncol * 64 + 16 * half + k0];
        acc = __builtin_amdgcn_wmma_f32_16x16x32_f16(
            /*neg_a=*/false, afrag, /*neg_b=*/false, bfrag,
            /*c_mod=*/(short)0, acc, /*reuse_a=*/false, /*reuse_b=*/false);
    }
    // C/D layout: VGPR r -> M = r + 8*half, N = lane&15 (+tile offset)
    for (int r = 0; r < 8; ++r) {
        int m = r + 8 * half;
        x1[((size_t)(nbase + m) * BB + b) * CC + ncol] = acc[r];
    }
}

// ---------------------------------------------------------------- 2) node degree
__global__ void k_deg(const int* __restrict__ node_idx, float* __restrict__ Dn) {
    int e = blockIdx.x * blockDim.x + threadIdx.x;
    if (e < EE) atomicAdd(&Dn[node_idx[e]], 1.0f);
}

// ---------------------------------------------------------------- 3) per-hedge sums + s + Binv
// block per hyperedge, 128 threads = one (b,c) lane each; hedge_idx sorted ->
// contiguous run, no atomics.
__global__ void k_edgesum(const float* __restrict__ x1, const int* __restrict__ node_idx,
                          const int* __restrict__ hedge_idx, const float* __restrict__ att,
                          float* __restrict__ esum, float* __restrict__ s,
                          float* __restrict__ Binv) {
    const int h = blockIdx.x;
    const int tid = threadIdx.x;
    int start, end;
    edge_range(hedge_idx, h, start, end);

    float acc = 0.0f;
    for (int e = start; e < end; ++e)
        acc += x1[(size_t)node_idx[e] * BC + tid];
    esum[(size_t)h * BC + tid] = acc;

    // s[h,b] = sum_c esum[h,b,c] * att[C+c]
    __shared__ float red[128];
    red[tid] = acc * att[CC + (tid & 63)];
    __syncthreads();
    for (int step = 32; step >= 1; step >>= 1) {
        if ((tid & 63) < step) red[tid] += red[tid + step];
        __syncthreads();
    }
    if ((tid & 63) == 0) s[h * BB + (tid >> 6)] = red[tid];
    if (tid == 0) Binv[h] = (end > start) ? 1.0f / (float)(end - start) : 0.0f;
}

// ---------------------------------------------------------------- 4) p[n,b] = x1[n,b,:].att[:C]
__global__ void k_pnode(const float* __restrict__ x1, const float* __restrict__ att,
                        float* __restrict__ p) {
    const int n = blockIdx.x;
    const int tid = threadIdx.x;
    __shared__ float red[128];
    red[tid] = x1[(size_t)n * BC + tid] * att[tid & 63];
    __syncthreads();
    for (int step = 32; step >= 1; step >>= 1) {
        if ((tid & 63) < step) red[tid] += red[tid + step];
        __syncthreads();
    }
    if ((tid & 63) == 0) p[n * BB + (tid >> 6)] = red[tid];
}

// ---------------------------------------------------------------- 5) logits + segment max
__global__ void k_alpha(const int* __restrict__ node_idx, const int* __restrict__ hedge_idx,
                        const float* __restrict__ p, const float* __restrict__ s,
                        float* __restrict__ alpha, unsigned* __restrict__ keymax) {
    int idx = blockIdx.x * blockDim.x + threadIdx.x;
    if (idx >= EE * BB) return;
    int e = idx >> 1, b = idx & 1;
    int n = node_idx[e], h = hedge_idx[e];
    float a = p[n * BB + b] + s[h * BB + b];
    a = (a >= 0.0f) ? a : 0.2f * a;            // leaky_relu(0.2)
    alpha[idx] = a;
    atomicMax(&keymax[n * BB + b], encf(a));
}

// ---------------------------------------------------------------- 6) exp + segment sum
__global__ void k_exp(const int* __restrict__ node_idx, float* __restrict__ alpha,
                      const unsigned* __restrict__ keymax, float* __restrict__ denom) {
    int idx = blockIdx.x * blockDim.x + threadIdx.x;
    if (idx >= EE * BB) return;
    int e = idx >> 1, b = idx & 1;
    int n = node_idx[e];
    float ex = __expf(alpha[idx] - decf(keymax[n * BB + b]));
    alpha[idx] = ex;
    atomicAdd(&denom[n * BB + b], ex);
}

// ---------------------------------------------------------------- 7) normalize
__global__ void k_norm(const int* __restrict__ node_idx, float* __restrict__ alpha,
                       const float* __restrict__ denom) {
    int idx = blockIdx.x * blockDim.x + threadIdx.x;
    if (idx >= EE * BB) return;
    int e = idx >> 1, b = idx & 1;
    alpha[idx] = alpha[idx] / denom[node_idx[e] * BB + b];
}

// ---------------------------------------------------------------- 8) pass 1: nodes -> hedges
__global__ void k_pass1(const float* __restrict__ x1, const int* __restrict__ node_idx,
                        const int* __restrict__ hedge_idx, const float* __restrict__ alpha,
                        const float* __restrict__ Binv, float* __restrict__ m) {
    const int h = blockIdx.x;
    const int tid = threadIdx.x;
    const int b = tid >> 6;
    int start, end;
    edge_range(hedge_idx, h, start, end);
    float acc = 0.0f;
    for (int e = start; e < end; ++e)
        acc += alpha[e * BB + b] * x1[(size_t)node_idx[e] * BC + tid];
    m[(size_t)h * BC + tid] = Binv[h] * acc;
}

// ---------------------------------------------------------------- 9) pass 2: hedges -> nodes (scatter into [B,N,C])
__global__ void k_pass2(const float* __restrict__ m, const int* __restrict__ node_idx,
                        const int* __restrict__ hedge_idx, const float* __restrict__ alpha,
                        const float* __restrict__ Dn, float* __restrict__ out) {
    const int e = blockIdx.x;
    const int tid = threadIdx.x;
    const int b = tid >> 6, c = tid & 63;
    int n = node_idx[e], h = hedge_idx[e];
    float coef = Dn[n] * alpha[e * BB + b];
    atomicAdd(&out[((size_t)b * NN + n) * CC + c], coef * m[(size_t)h * BC + tid]);
}

// ---------------------------------------------------------------- launch
extern "C" void kernel_launch(void* const* d_in, const int* in_sizes, int n_in,
                              void* d_out, int out_size, void* d_ws, size_t ws_size,
                              hipStream_t stream) {
    const float* x        = (const float*)d_in[0];   // [B,N,C]
    const float* weight   = (const float*)d_in[1];   // [C,C]
    const float* att      = (const float*)d_in[2];   // [2C]
    const int*   node_idx = (const int*)d_in[3];     // [E]
    const int*   hedge_idx= (const int*)d_in[4];     // [E] sorted
    float* out = (float*)d_out;                      // [B,N,C]

    // carve workspace (~35.4 MB total)
    char* base = (char*)d_ws;
    size_t off = 0;
    auto carve = [&](size_t bytes) -> char* {
        char* p = base + off;
        off = (off + bytes + 255) & ~(size_t)255;
        return p;
    };
    float*    x1     = (float*)   carve((size_t)NN * BC * 4);   // 25.6 MB
    float*    esum   = (float*)   carve((size_t)HH * BC * 4);   // 2.56 MB
    float*    mbuf   = (float*)   carve((size_t)HH * BC * 4);   // 2.56 MB
    float*    alpha  = (float*)   carve((size_t)EE * BB * 4);   // 3.2 MB
    float*    p      = (float*)   carve((size_t)NN * BB * 4);
    float*    s      = (float*)   carve((size_t)HH * BB * 4);
    unsigned* keymax = (unsigned*)carve((size_t)NN * BB * 4);
    float*    denom  = (float*)   carve((size_t)NN * BB * 4);
    float*    Dn     = (float*)   carve((size_t)NN * 4);
    float*    Binv   = (float*)   carve((size_t)HH * 4);
    (void)ws_size; (void)in_sizes; (void)n_in; (void)out_size;

    // zero accumulation targets (every call: graph replays must be deterministic)
    k_zero<<<(BB * NN * CC + 255) / 256, 256, 0, stream>>>(out, BB * NN * CC);
    k_zero<<<(NN + 255) / 256, 256, 0, stream>>>(Dn, NN);
    k_zero<<<(NN * BB + 255) / 256, 256, 0, stream>>>(denom, NN * BB);
    k_zero<<<(NN * BB + 255) / 256, 256, 0, stream>>>((float*)keymax, NN * BB);

    // 1) projection GEMM (WMMA), output node-major [N,B,C]
    k_gemm<<<dim3(NN / 16, BB), 128, 0, stream>>>(x, weight, x1);
    // 2) node degrees
    k_deg<<<(EE + 255) / 256, 256, 0, stream>>>(node_idx, Dn);
    // 3) per-hyperedge feature sums + attention projection + 1/De
    k_edgesum<<<HH, 128, 0, stream>>>(x1, node_idx, hedge_idx, att, esum, s, Binv);
    // 4) per-node attention projection
    k_pnode<<<NN, 128, 0, stream>>>(x1, att, p);
    // 5-7) segment softmax over nodes
    k_alpha<<<(EE * BB + 255) / 256, 256, 0, stream>>>(node_idx, hedge_idx, p, s, alpha, keymax);
    k_exp  <<<(EE * BB + 255) / 256, 256, 0, stream>>>(node_idx, alpha, keymax, denom);
    k_norm <<<(EE * BB + 255) / 256, 256, 0, stream>>>(node_idx, alpha, denom);
    // 8) nodes -> hyperedges
    k_pass1<<<HH, 128, 0, stream>>>(x1, node_idx, hedge_idx, alpha, Binv, mbuf);
    // 9) hyperedges -> nodes, scattered directly into [B,N,C] output
    k_pass2<<<EE, 128, 0, stream>>>(mbuf, node_idx, hedge_idx, alpha, Dn, out);
}
